// GNNEncoder_1202590843452
// MI455X (gfx1250) — matrix operand order repaired and checked by reading, not compiled, not run on hardware
//
#include <hip/hip_runtime.h>

#define NN 50000          // nodes
#define NE 800000         // raw edges
#define NL 850000         // edges + self loops
#define HIDC 64           // per-head channels
#define GG 512            // graphs
#define NEG_SLOPE 0.2f

typedef __attribute__((ext_vector_type(16))) _Float16 v16h;
typedef __attribute__((ext_vector_type(8)))  _Float16 v8h;
typedef __attribute__((ext_vector_type(8)))  float    v8f;

// ---------------------------------------------------------------------------
// float atomic max via monotonic int mapping (lowers to global_atomic_max_i32 /
// global_atomic_min_u32); valid with -inf init.
// ---------------------------------------------------------------------------
__device__ __forceinline__ void atomicMaxF(float* addr, float val) {
  if (val >= 0.0f)
    atomicMax((int*)addr, __float_as_int(val));
  else
    atomicMin((unsigned int*)addr, __float_as_uint(val));
}

__global__ void fill_f32(float* __restrict__ p, float v, int n) {
  int i = blockIdx.x * blockDim.x + threadIdx.x;
  if (i < n) p[i] = v;
}

__global__ void cvt_f16(const float* __restrict__ s, _Float16* __restrict__ d, int n) {
  int i = blockIdx.x * blockDim.x + threadIdx.x;
  if (i < n) d[i] = (_Float16)s[i];
}

// Wt[o*K+k] = (f16) W[k*OUT+o]   (so B-tiles read contiguous K per output col)
__global__ void wtrans_f16(const float* __restrict__ W, _Float16* __restrict__ Wt,
                           int K, int OUT) {
  int i = blockIdx.x * blockDim.x + threadIdx.x;
  if (i >= K * OUT) return;
  int k = i / OUT, o = i - k * OUT;
  Wt[(size_t)o * K + k] = (_Float16)W[i];
}

// ---------------------------------------------------------------------------
// WMMA GEMM: C[M,OUT] = A[M,K] * Bt^T, Bt[OUT,K] pre-transposed.
// One wave computes a 16x64 C strip: 4 column tiles sharing one A fragment
// per K-step (4 v_wmma_f32_16x16x32_f16 per 10 b128 loads, A traffic /4).
// A lane layout (ISA 7.12.2, 16-bit A 16x32): lanes 0-15: K {0..7,16..23},
// lanes 16-31: K {8..15,24..31}.  B lane layout: same K order as A
// (lanes 0-15: K {0..7,16..23}, lanes 16-31: K {8..15,24..31}).
// ---------------------------------------------------------------------------
__global__ void wmma_gemm_f16(const _Float16* __restrict__ A,
                              const _Float16* __restrict__ Bt,
                              float* __restrict__ C,
                              int M, int K, int OUT) {
  int wave = blockIdx.x * (blockDim.x >> 5) + (threadIdx.x >> 5);
  int lane = threadIdx.x & 31;
  int colGroups = OUT >> 6;                    // 64 output cols per wave
  int nWaves = (M >> 4) * colGroups;
  if (wave >= nWaves) return;
  int rowTile = wave / colGroups;
  int colGrp  = wave - rowTile * colGroups;
  int hi  = lane >> 4;                         // 0 or 1
  int l15 = lane & 15;
  size_t arow = (size_t)(rowTile * 16 + l15) * K;
  size_t bbase = (size_t)(colGrp * 64 + l15) * K;

  v8f acc[4] = {};
  for (int k0 = 0; k0 < K; k0 += 32) {
    const v8h* ap = (const v8h*)(A + arow + k0 + hi * 8);
    v8h a0 = ap[0], a1 = ap[2];                // K = sel..sel+7, sel+16..sel+23
    v16h a;
#pragma unroll
    for (int i = 0; i < 8; ++i) { a[i] = a0[i]; a[i + 8] = a1[i]; }
#pragma unroll
    for (int t = 0; t < 4; ++t) {
      const v8h* bp = (const v8h*)(Bt + bbase + (size_t)t * 16 * K + k0 + hi * 8);
      v8h b0 = bp[0], b1 = bp[2];              // K = sel..sel+7, sel+16..sel+23
      v16h b;
#pragma unroll
      for (int i = 0; i < 8; ++i) { b[i] = b0[i]; b[i + 8] = b1[i]; }
      acc[t] = __builtin_amdgcn_wmma_f32_16x16x32_f16(
          /*neg_a=*/false, a, /*neg_b=*/false, b,
          /*c_mod=*/(short)0, acc[t], /*reuse_a=*/false, /*reuse_b=*/false);
    }
  }
  // C/D layout: VGPR i -> row rowTile*16 + i + hi*8, col tile base + l15
  float* cp = C + (size_t)(rowTile * 16 + hi * 8) * OUT + colGrp * 64 + l15;
#pragma unroll
  for (int t = 0; t < 4; ++t)
#pragma unroll
    for (int i = 0; i < 8; ++i) cp[(size_t)i * OUT + t * 16] = acc[t][i];
}

// a_s[n,h] = sum_c z[n,h,c]*att_src[h,c]; also resets softmax state
__global__ void attn_coeff(const float* __restrict__ Z,
                           const float* __restrict__ attS,
                           const float* __restrict__ attD,
                           float* __restrict__ AS, float* __restrict__ AD,
                           float* __restrict__ EMAX, float* __restrict__ DEN, int H) {
  int i = blockIdx.x * blockDim.x + threadIdx.x;
  if (i >= NN * H) return;
  int node = i / H, h = i - node * H;
  const float* z = Z + (size_t)node * H * HIDC + h * HIDC;
  float s = 0.f, d = 0.f;
  for (int c = 0; c < HIDC; ++c) {
    float v = z[c];
    s += v * attS[h * HIDC + c];
    d += v * attD[h * HIDC + c];
  }
  AS[i] = s; AD[i] = d;
  EMAX[i] = -__builtin_inff();
  DEN[i] = 0.f;
}

// pass 1: leaky-relu logit per (edge, head); atomic max into EMAX[dst,h]
__global__ void edge_logits(const int* __restrict__ ei,
                            const float* __restrict__ AS, const float* __restrict__ AD,
                            float* __restrict__ EB, float* __restrict__ EMAX, int H) {
  int idx = blockIdx.x * blockDim.x + threadIdx.x;
  if (idx >= NL * H) return;
  int e = idx / H, h = idx - e * H;
  int src = (e < NE) ? ei[e]      : (e - NE);
  int dst = (e < NE) ? ei[NE + e] : (e - NE);
  float a = AS[src * H + h] + AD[dst * H + h];
  float lr = (a > 0.f) ? a : NEG_SLOPE * a;
  EB[idx] = lr;
  atomicMaxF(&EMAX[dst * H + h], lr);
}

// pass 2: e_exp = exp(e - max); atomic add into DEN[dst,h]
__global__ void edge_exp(const int* __restrict__ ei,
                         float* __restrict__ EB, const float* __restrict__ EMAX,
                         float* __restrict__ DEN, int H) {
  int idx = blockIdx.x * blockDim.x + threadIdx.x;
  if (idx >= NL * H) return;
  int e = idx / H, h = idx - e * H;
  int dst = (e < NE) ? ei[NE + e] : (e - NE);
  float ex = expf(EB[idx] - EMAX[dst * H + h]);
  EB[idx] = ex;
  atomicAdd(&DEN[dst * H + h], ex);
}

// pass 3: out[dst, :] += z[src, :] * alpha  (thread per (edge, channel))
__global__ void edge_scatter(const int* __restrict__ ei,
                             const float* __restrict__ Z,
                             const float* __restrict__ EB, const float* __restrict__ DEN,
                             float* __restrict__ OUTB, int H) {
  long long idx = (long long)blockIdx.x * blockDim.x + threadIdx.x;
  int HC = H * HIDC;
  long long total = (long long)NL * HC;
  if (idx >= total) return;
  int e = (int)(idx / HC);
  int j = (int)(idx - (long long)e * HC);
  int h = j >> 6;
  int src = (e < NE) ? ei[e]      : (e - NE);
  int dst = (e < NE) ? ei[NE + e] : (e - NE);
  float alpha = EB[(size_t)e * H + h] / (DEN[dst * H + h] + 1e-16f);
  atomicAdd(&OUTB[(size_t)dst * HC + j], Z[(size_t)src * HC + j] * alpha);
}

// bias + ELU, write next layer's f16 input
__global__ void bias_elu_h16(const float* __restrict__ OUTB, const float* __restrict__ bias,
                             _Float16* __restrict__ H16, int HC) {
  int idx = blockIdx.x * blockDim.x + threadIdx.x;
  if (idx >= NN * HC) return;
  int j = idx % HC;
  float v = OUTB[idx] + bias[j];
  v = (v > 0.f) ? v : (expf(v) - 1.f);
  H16[idx] = (_Float16)v;
}

// final: bias + segment max pool over batch -> out[G, 64]
__global__ void pool_max(const float* __restrict__ OUTB, const float* __restrict__ bias,
                         const int* __restrict__ batch, float* __restrict__ out) {
  int idx = blockIdx.x * blockDim.x + threadIdx.x;
  if (idx >= NN * HIDC) return;
  int node = idx >> 6, c = idx & 63;
  atomicMaxF(&out[(size_t)batch[node] * HIDC + c], OUTB[idx] + bias[c]);
}

// ---------------------------------------------------------------------------
extern "C" void kernel_launch(void* const* d_in, const int* in_sizes, int n_in,
                              void* d_out, int out_size, void* d_ws, size_t ws_size,
                              hipStream_t stream) {
  (void)in_sizes; (void)n_in; (void)out_size; (void)ws_size;
  const float* x    = (const float*)d_in[0];
  const int*   ei   = (const int*)d_in[1];
  const int*   batch= (const int*)d_in[2];
  const float* W0   = (const float*)d_in[3];
  const float* aS0  = (const float*)d_in[4];
  const float* aD0  = (const float*)d_in[5];
  const float* b0   = (const float*)d_in[6];
  const float* W1   = (const float*)d_in[7];
  const float* aS1  = (const float*)d_in[8];
  const float* aD1  = (const float*)d_in[9];
  const float* b1   = (const float*)d_in[10];
  const float* W2   = (const float*)d_in[11];
  const float* aS2  = (const float*)d_in[12];
  const float* aD2  = (const float*)d_in[13];
  const float* b2   = (const float*)d_in[14];
  float* out = (float*)d_out;

  char* ws = (char*)d_ws;
  size_t off = 0;
  auto alloc = [&](size_t bytes) -> void* {
    void* p = ws + off;
    off += (bytes + 255) & ~(size_t)255;
    return p;
  };
  _Float16* Wt0  = (_Float16*)alloc((size_t)64  * 256 * 2);
  _Float16* Wt1  = (_Float16*)alloc((size_t)256 * 256 * 2);
  _Float16* Wt2  = (_Float16*)alloc((size_t)256 * 64  * 2);
  _Float16* XF16 = (_Float16*)alloc((size_t)NN * 256 * 2);
  float*    Z    = (float*)alloc((size_t)NN * 256 * 4);
  float*    OUTB = (float*)alloc((size_t)NN * 256 * 4);
  float*    AS   = (float*)alloc((size_t)NN * 4 * 4);
  float*    AD   = (float*)alloc((size_t)NN * 4 * 4);
  float*    EMAX = (float*)alloc((size_t)NN * 4 * 4);
  float*    DEN  = (float*)alloc((size_t)NN * 4 * 4);
  float*    EBUF = (float*)alloc((size_t)NL * 4 * 4);

  const int T = 256;
  auto blk = [](long long n) { return (int)((n + 255) / 256); };

  // weight transpose + f16 convert (cheap, deterministic each call)
  wtrans_f16<<<blk(64 * 256),  T, 0, stream>>>(W0, Wt0, 64, 256);
  wtrans_f16<<<blk(256 * 256), T, 0, stream>>>(W1, Wt1, 256, 256);
  wtrans_f16<<<blk(256 * 64),  T, 0, stream>>>(W2, Wt2, 256, 64);
  cvt_f16<<<blk((long long)NN * 64), T, 0, stream>>>(x, XF16, NN * 64);

  struct Layer { const _Float16* Wt; const float *as, *ad, *b; int K, OUT, H; };
  Layer L[3] = {
    { Wt0, aS0, aD0, b0, 64,  256, 4 },
    { Wt1, aS1, aD1, b1, 256, 256, 4 },
    { Wt2, aS2, aD2, b2, 256, 64,  1 },
  };

  for (int li = 0; li < 3; ++li) {
    const Layer& P = L[li];
    int HC = P.OUT;
    // GEMM: one wave per 16x64 C strip (4 wmma tiles), 8 waves / block
    int gwaves = (NN / 16) * (P.OUT / 64);
    wmma_gemm_f16<<<(gwaves + 7) / 8, T, 0, stream>>>(XF16, P.Wt, Z, NN, P.K, P.OUT);
    // attention coefficients + softmax state init (fused)
    attn_coeff<<<blk((long long)NN * P.H), T, 0, stream>>>(Z, P.as, P.ad, AS, AD,
                                                           EMAX, DEN, P.H);
    fill_f32<<<blk((long long)NN * HC), T, 0, stream>>>(OUTB, 0.f, NN * HC);
    // segment softmax + message scatter
    edge_logits <<<blk((long long)NL * P.H), T, 0, stream>>>(ei, AS, AD, EBUF, EMAX, P.H);
    edge_exp    <<<blk((long long)NL * P.H), T, 0, stream>>>(ei, EBUF, EMAX, DEN, P.H);
    edge_scatter<<<blk((long long)NL * HC),  T, 0, stream>>>(ei, Z, EBUF, DEN, OUTB, P.H);
    if (li < 2) {
      bias_elu_h16<<<blk((long long)NN * HC), T, 0, stream>>>(OUTB, P.b, XF16, HC);
    } else {
      fill_f32<<<blk(GG * HIDC), T, 0, stream>>>(out, -__builtin_inff(), GG * HIDC);
      pool_max<<<blk((long long)NN * HIDC), T, 0, stream>>>(OUTB, P.b, batch, out);
    }
  }
}
